// BarycentricPooling_22660247453772
// MI455X (gfx1250) — matrix operand order repaired and checked by reading, not compiled, not run on hardware
//
#include <hip/hip_runtime.h>
#include <math.h>

// ---------------------------------------------------------------------------
// BarycentricPooling (Sinkhorn OT histograms + segment mean), MI455X / gfx1250
//   N=20000 nodes, S=32 samples, D=256 dims, K=64 codes, B=256 graphs
//   Roofline: 655 MB X read once (~28us @ 23.3 TB/s) dominates; GEMM on WMMA,
//   Sinkhorn fully in registers (online logsumexp, shuffle broadcasts).
// ---------------------------------------------------------------------------

typedef __attribute__((ext_vector_type(16))) _Float16 v16h;
typedef __attribute__((ext_vector_type(8)))  float    v8f;
typedef __attribute__((ext_vector_type(4)))  float    f32x4;

namespace cfg {
constexpr int S = 32;          // samples per node
constexpr int D = 256;         // feature dim
constexpr int K = 64;          // codebook size
constexpr int NB = 256;        // number of graphs
constexpr int WAVES = 8;       // nodes per block (1 node per wave32)
constexpr int CPITCH = 65;     // LDS pitch for 32x64 C tile (conflict-free)
constexpr int ITERS = 31;      // 30 detached + 1 differentiable update
constexpr float EPS = 0.1f;
constexpr float INV_EPS = 10.0f;
constexpr float LOG_A = -3.46573590279972655f;  // -log(32)
}

// Load one 16x32 f16 A-tile row-slice for this lane per the CDNA5 WMMA A
// layout (lanes 0-15: K={0..7,16..23}; lanes 16-31: K={8..15,24..31}),
// accumulating the f32 sum of squares as a side product.
__device__ inline v16h load_a_tile(const float* __restrict__ rb, float& sq) {
  f32x4 p0 = *(const f32x4*)(rb);        // K+0..3   (relative)
  f32x4 p1 = *(const f32x4*)(rb + 4);    // K+4..7
  f32x4 p2 = *(const f32x4*)(rb + 16);   // K+16..19
  f32x4 p3 = *(const f32x4*)(rb + 20);   // K+20..23
  sq += p0.x*p0.x + p0.y*p0.y + p0.z*p0.z + p0.w*p0.w
      + p1.x*p1.x + p1.y*p1.y + p1.z*p1.z + p1.w*p1.w
      + p2.x*p2.x + p2.y*p2.y + p2.z*p2.z + p2.w*p2.w
      + p3.x*p3.x + p3.y*p3.y + p3.z*p3.z + p3.w*p3.w;
  v16h a;
  a[0]=(_Float16)p0.x;  a[1]=(_Float16)p0.y;  a[2]=(_Float16)p0.z;  a[3]=(_Float16)p0.w;
  a[4]=(_Float16)p1.x;  a[5]=(_Float16)p1.y;  a[6]=(_Float16)p1.z;  a[7]=(_Float16)p1.w;
  a[8]=(_Float16)p2.x;  a[9]=(_Float16)p2.y;  a[10]=(_Float16)p2.z; a[11]=(_Float16)p2.w;
  a[12]=(_Float16)p3.x; a[13]=(_Float16)p3.y; a[14]=(_Float16)p3.z; a[15]=(_Float16)p3.w;
  return a;
}

__global__ __launch_bounds__(256)
void ot_hist_kernel(const float* __restrict__ X,
                    const int*   __restrict__ batch_idx,
                    const float* __restrict__ codebook,
                    const float* __restrict__ log_prior,
                    float* __restrict__ sums,    // [NB*K]
                    float* __restrict__ counts,  // [NB]
                    int N) {
  using namespace cfg;
  // LDS: codebook in f16 B-tile layout (32KB) + per-wave C tiles (~65KB)
  __shared__ __align__(32) _Float16 ldsB[4 * 8 * 32 * 16];  // [nt][kc][lane][e]
  __shared__ float ldsC[WAVES][S * CPITCH];
  __shared__ float y2s[K];
  __shared__ float lbs[K];
  __shared__ float lse_s;

  const int t    = threadIdx.x;
  const int lane = t & 31;
  const int w    = t >> 5;
  const int node = blockIdx.x * WAVES + w;
  const bool active = node < N;

  const float* xbase = X + (size_t)(active ? node : 0) * S * D;

  // Prefetch this wave's X rows into cache (global_prefetch_b8).
  if (active) __builtin_prefetch(xbase + (size_t)lane * D, 0, 1);

  // ---------------- per-block staging: log_softmax(prior), y2, B tiles -----
  if (t == 0) {
    float m = -INFINITY;
    for (int k = 0; k < K; ++k) m = fmaxf(m, log_prior[k]);
    float ss = 0.0f;
    for (int k = 0; k < K; ++k) ss += __expf(log_prior[k] - m);
    lse_s = m + __logf(ss);
  }
  if (t < K) {
    float acc = 0.0f;
    const float* cp = codebook + t * D;
    for (int d = 0; d < D; d += 4) {
      f32x4 v = *(const f32x4*)(cp + d);
      acc += v.x*v.x + v.y*v.y + v.z*v.z + v.w*v.w;
    }
    y2s[t] = acc;
  }
  // Stage codebook as f16 in CDNA5 B-matrix layout:
  // lane<16 holds K={0..7,16..23}, lane>=16 holds K={8..15,24..31}; N = lane&15.
  for (int idx = t; idx < 4 * 8 * 32 * 16; idx += 256) {
    int e  = idx & 15;
    int bl = (idx >> 4) & 31;
    int kc = (idx >> 9) & 7;
    int nt = idx >> 12;
    int col = nt * 16 + (bl & 15);
    int d   = kc * 32 + (e & 7) + ((e & 8) ? 16 : 0) + ((bl >= 16) ? 8 : 0);
    ldsB[idx] = (_Float16)codebook[col * D + d];
  }
  __syncthreads();
  if (t < K) lbs[t] = log_prior[t] - lse_s;
  __syncthreads();

  // ---------------- GEMM: dot = X @ codebook^T via v_wmma_f32_16x16x32_f16 -
  const int hi = (lane >= 16) ? 1 : 0;
  const int lo = lane & 15;

  v8f acc[2][4];
#pragma unroll
  for (int mt = 0; mt < 2; ++mt)
#pragma unroll
    for (int nt = 0; nt < 4; ++nt) {
      v8f z = {0.f, 0.f, 0.f, 0.f, 0.f, 0.f, 0.f, 0.f};
      acc[mt][nt] = z;
    }

  float sq0 = 0.0f, sq1 = 0.0f;  // per-lane partial sum-of-squares (rows lo, 16+lo)
#pragma unroll
  for (int kc = 0; kc < 8; ++kc) {
    const float* r0 = xbase + (size_t)lo        * D + kc * 32 + hi * 8;
    const float* r1 = xbase + (size_t)(16 + lo) * D + kc * 32 + hi * 8;
    v16h A0 = load_a_tile(r0, sq0);
    v16h A1 = load_a_tile(r1, sq1);
#pragma unroll
    for (int nt = 0; nt < 4; ++nt) {
      v16h b = *(const v16h*)&ldsB[((nt * 8 + kc) * 32 + lane) * 16];
      acc[0][nt] = __builtin_amdgcn_wmma_f32_16x16x32_f16(
          false, A0, false, b, (short)0, acc[0][nt], false, false);
      acc[1][nt] = __builtin_amdgcn_wmma_f32_16x16x32_f16(
          false, A1, false, b, (short)0, acc[1][nt], false, false);
    }
  }

  // Full-row x2 by pairing lanes (L, L^16) covering disjoint K halves.
  float x20 = sq0 + __shfl_xor(sq0, 16, 32);  // x2 of row lo      (M-tile 0)
  float x21 = sq1 + __shfl_xor(sq1, 16, 32);  // x2 of row 16+lo   (M-tile 1)

  // ---------------- C = max(0, x2 + y2 - 2*dot) -> LDS (pitch 65) ----------
  float* Cw = &ldsC[w][0];
#pragma unroll
  for (int mt = 0; mt < 2; ++mt) {
    float xsrc = (mt == 0) ? x20 : x21;
#pragma unroll
    for (int r = 0; r < 8; ++r) {
      int rowInTile = r + hi * 8;               // D-layout: M = r + 8*(lane>=16)
      float x2v = __shfl(xsrc, rowInTile, 32);  // x2 lives in lane == rowInTile
      int srow = mt * 16 + rowInTile;
#pragma unroll
      for (int nt = 0; nt < 4; ++nt) {
        int kcol = nt * 16 + lo;                // D-layout: N = lane&15
        float c = fmaxf(x2v + y2s[kcol] - 2.0f * acc[mt][nt][r], 0.0f);
        Cw[srow * CPITCH + kcol] = c;
      }
    }
  }
  __syncthreads();

  // Cache this lane's two C columns (k=lane, k=lane+32) in VGPRs: the whole
  // g-half of Sinkhorn then runs register-only.
  float cC0[32], cC1[32];
#pragma unroll
  for (int s = 0; s < 32; ++s) {
    cC0[s] = Cw[s * CPITCH + lane];
    cC1[s] = Cw[s * CPITCH + lane + 32];
  }
  const float lb0 = lbs[lane];
  const float lb1 = lbs[lane + 32];

  // ---------------- Sinkhorn: 31 log-domain updates, online logsumexp ------
  float f = 0.0f, g0 = 0.0f, g1 = 0.0f;
#pragma unroll 1
  for (int it = 0; it < ITERS; ++it) {
    float lg0 = lb0 + g0 * INV_EPS;  // lane owns (lb+g/eps) for k=lane, lane+32
    float lg1 = lb1 + g1 * INV_EPS;
    // f-update: lane = sample row s; stream C row from LDS (conflict-free).
    const float* crow = &Cw[lane * CPITCH];
    float m = -INFINITY, ss = 0.0f;
#pragma unroll
    for (int k = 0; k < K; ++k) {
      float lg = __shfl((k < 32) ? lg0 : lg1, k & 31, 32);
      float z  = lg - crow[k] * INV_EPS;
      float mn = fmaxf(m, z);
      ss = ss * __expf(m - mn) + __expf(z - mn);
      m  = mn;
    }
    f = -EPS * (m + __logf(ss));
    // g-update: register-only (column cache + shuffle broadcast of f).
    float m0 = -INFINITY, s0 = 0.0f, m1 = -INFINITY, s1 = 0.0f;
#pragma unroll
    for (int s = 0; s < 32; ++s) {
      float fs   = __shfl(f, s, 32);
      float base = LOG_A + fs * INV_EPS;
      float z0 = base - cC0[s] * INV_EPS;
      float z1 = base - cC1[s] * INV_EPS;
      float mn0 = fmaxf(m0, z0);
      s0 = s0 * __expf(m0 - mn0) + __expf(z0 - mn0); m0 = mn0;
      float mn1 = fmaxf(m1, z1);
      s1 = s1 * __expf(m1 - mn1) + __expf(z1 - mn1); m1 = mn1;
    }
    g0 = -EPS * (m0 + __logf(s0));
    g1 = -EPS * (m1 + __logf(s1));
  }

  // ---------------- histogram: hist[k] = exp(logsumexp_s log_pi) -----------
  float m0 = -INFINITY, s0 = 0.0f, m1 = -INFINITY, s1 = 0.0f;
#pragma unroll
  for (int s = 0; s < 32; ++s) {
    float fs = __shfl(f, s, 32);
    float z0 = (fs + g0 - cC0[s]) * INV_EPS + LOG_A + lb0;
    float z1 = (fs + g1 - cC1[s]) * INV_EPS + LOG_A + lb1;
    float mn0 = fmaxf(m0, z0);
    s0 = s0 * __expf(m0 - mn0) + __expf(z0 - mn0); m0 = mn0;
    float mn1 = fmaxf(m1, z1);
    s1 = s1 * __expf(m1 - mn1) + __expf(z1 - mn1); m1 = mn1;
  }
  float h0 = __expf(m0) * s0;
  float h1 = __expf(m1) * s1;

  // Normalize over k within the wave, then scatter into per-graph sums.
  float tot = h0 + h1;
#pragma unroll
  for (int off = 16; off > 0; off >>= 1) tot += __shfl_xor(tot, off, 32);
  float inv = 1.0f / (tot + 1e-12f);

  if (active) {
    int b = batch_idx[node];
    atomicAdd(&sums[b * K + lane],      h0 * inv);
    atomicAdd(&sums[b * K + lane + 32], h1 * inv);
    if (lane == 0) atomicAdd(&counts[b], 1.0f);
  }
}

__global__ void zero_ws_kernel(float* __restrict__ p, int n) {
  int i = blockIdx.x * blockDim.x + threadIdx.x;
  if (i < n) p[i] = 0.0f;
}

__global__ void finalize_kernel(const float* __restrict__ sums,
                                const float* __restrict__ counts,
                                const float* __restrict__ log_prior,
                                float* __restrict__ out) {
  using namespace cfg;
  __shared__ float lse;
  int b = blockIdx.x;
  int k = threadIdx.x;  // K=64 threads
  if (k == 0) {
    float m = -INFINITY;
    for (int j = 0; j < K; ++j) m = fmaxf(m, log_prior[j]);
    float ss = 0.0f;
    for (int j = 0; j < K; ++j) ss += __expf(log_prior[j] - m);
    lse = m + __logf(ss);
  }
  __syncthreads();
  float c = counts[b];
  out[b * K + k] = (c > 0.0f) ? (sums[b * K + k] / c)
                              : __expf(log_prior[k] - lse);
}

extern "C" void kernel_launch(void* const* d_in, const int* in_sizes, int n_in,
                              void* d_out, int out_size, void* d_ws, size_t ws_size,
                              hipStream_t stream) {
  using namespace cfg;
  const float* X    = (const float*)d_in[0];  // [N,S,D] f32
  const int*   bidx = (const int*)d_in[1];    // [N] i32
  const float* cb   = (const float*)d_in[2];  // [K,D] f32
  const float* lp   = (const float*)d_in[3];  // [K] f32
  float* out = (float*)d_out;                 // [NB,K] f32

  const int N = in_sizes[0] / (S * D);        // 20000

  float* sums   = (float*)d_ws;               // [NB*K]
  float* counts = sums + NB * K;              // [NB]
  const int wsn = NB * K + NB;

  zero_ws_kernel<<<(wsn + 255) / 256, 256, 0, stream>>>(sums, wsn);

  const int nblocks = (N + WAVES - 1) / WAVES;  // 2500
  ot_hist_kernel<<<nblocks, 256, 0, stream>>>(X, bidx, cb, lp, sums, counts, N);

  finalize_kernel<<<NB, K, 0, stream>>>(sums, counts, lp, out);
}
